// MixTransformer_61400852464111
// MI455X (gfx1250) — matrix operand order repaired and checked by reading, not compiled, not run on hardware
//
#include <hip/hip_runtime.h>

typedef unsigned short u16;
typedef __attribute__((ext_vector_type(16))) __bf16 v16bf;
typedef __attribute__((ext_vector_type(8)))  float  v8f;

#ifndef __has_builtin
#define __has_builtin(x) 0
#endif
#if __has_builtin(__builtin_amdgcn_tensor_load_to_lds) && \
    __has_builtin(__builtin_amdgcn_s_wait_tensorcnt)
#define HAVE_TDM 1
#else
#define HAVE_TDM 0
#endif

#define LDS_AS __attribute__((address_space(3)))

// ---------------- constants ----------------
#define Bsz   2
#define Sq    1024
#define Dm    2048
#define Hn    16
#define KVHn  8
#define HDn   128
#define DFFn  8192
#define En    8
#define Rr    16
#define Mtok  (Bsz * Sq)       // 2048 tokens
#define SCALING 2.0f           // 32/16

// ---------------- helpers ----------------
__device__ __forceinline__ u16 f2bf(float f) {
  union { float f; unsigned u; } x; x.f = f;
  unsigned r = x.u + 0x7FFFu + ((x.u >> 16) & 1u);
  return (u16)(r >> 16);
}
__device__ __forceinline__ float bf2f(u16 h) {
  union { unsigned u; float f; } x; x.u = ((unsigned)h) << 16;
  return x.f;
}

struct Frag {
  union { v16bf v; uint4 q[2]; };
};

__device__ __forceinline__ v8f wmma_bf16(const Frag& a, const Frag& b, v8f c) {
  // D = A(16x32 bf16) * B(32x16 bf16) + C(16x16 f32)
  return __builtin_amdgcn_wmma_f32_16x16x32_bf16(false, a.v, false, b.v,
                                                 (short)0, c, false, false);
}

#if HAVE_TDM
__device__ __forceinline__ unsigned lds_offset(void* p) {
  return (unsigned)(unsigned long long)(LDS_AS char*)p;
}
// 2D TDM tile load: tile 128 rows x 64 bf16, row stride = strideElems,
// LDS rows padded by 16B every 128B (pad_interval=32 DW, pad_amount=4 DW) -> 72-elem rows.
__device__ __forceinline__ void tdm_load_tile(const u16* gptr, unsigned ldsOff,
                                              int kRem, int mRem, int strideElems) {
  typedef unsigned v4u __attribute__((ext_vector_type(4)));
  typedef int      v8i __attribute__((ext_vector_type(8)));
  typedef int      v4i __attribute__((ext_vector_type(4)));
  unsigned long long ga = (unsigned long long)(size_t)gptr;
  v4u g0;
  g0[0] = 1u;                                        // count=1 (valid), user mode
  g0[1] = ldsOff;                                    // lds_addr (bytes)
  g0[2] = (unsigned)(ga & 0xffffffffu);              // global_addr[31:0]
  g0[3] = (unsigned)((ga >> 32) & 0x01ffffffu)       // global_addr[56:32]
        | (2u << 30);                                // type=2 ("image")
  unsigned td0 = (unsigned)kRem;                     // tensor_dim0 (elems)
  unsigned td1 = (unsigned)mRem;                     // tensor_dim1 (rows)
  unsigned long long st0 = (unsigned long long)strideElems;
  v8i g1;
  g1[0] = (int)((1u << 16)          // data_size = 2 bytes
              | (1u << 20)          // pad_enable
              | (4u << 22)          // pad_interval: 32 DWORDs (=128B)
              | (3u << 25));        // pad_amount: 4 DWORDs (=16B)
  g1[1] = (int)((td0 & 0xffffu) << 16);              // dim0 low16 (atomic addr = 0)
  g1[2] = (int)((td0 >> 16) | ((td1 & 0xffffu) << 16));
  g1[3] = (int)((td1 >> 16) | (64u << 16));          // tile_dim0 = 64 elems
  g1[4] = 128;                                       // tile_dim1 = 128 rows, tile_dim2 = 0
  g1[5] = (int)(st0 & 0xffffffffu);                  // dim0_stride low32
  g1[6] = (int)((st0 >> 32) & 0xffffu);              // dim0_stride high16
  g1[7] = 0;
  v4i z4 = {0, 0, 0, 0};
  v8i z8 = {0, 0, 0, 0, 0, 0, 0, 0};
  // 6-arg form (clang-23 / therock-10.0 lane)
  __builtin_amdgcn_tensor_load_to_lds(g0, g1, z4, z4, z8, 0);
}
#endif

// ---------------- f32 -> bf16 convert (grid-stride) ----------------
__global__ void cvt_bf16(const float* __restrict__ X, u16* __restrict__ Y, size_t n) {
  for (size_t i = (size_t)blockIdx.x * blockDim.x + threadIdx.x; i < n;
       i += (size_t)gridDim.x * blockDim.x)
    Y[i] = f2bf(X[i]);
}

__global__ void copy_f32(const float* __restrict__ X, float* __restrict__ Y, size_t n) {
  for (size_t i = (size_t)blockIdx.x * blockDim.x + threadIdx.x; i < n;
       i += (size_t)gridDim.x * blockDim.x)
    Y[i] = X[i];
}

// ---------------- RMSNorm -> bf16 ----------------
__global__ __launch_bounds__(256) void rmsnorm_bf16(
    const float* __restrict__ X, const float* __restrict__ W,
    u16* __restrict__ Y, int D) {
  __shared__ float red[8];
  const int m = blockIdx.x;
  const int tid = threadIdx.x;
  const float* x = X + (size_t)m * D;
  float ss = 0.f;
  for (int i = tid; i < D; i += 256) { float v = x[i]; ss += v * v; }
#pragma unroll
  for (int off = 16; off >= 1; off >>= 1) ss += __shfl_xor(ss, off, 32);
  if ((tid & 31) == 0) red[tid >> 5] = ss;
  __syncthreads();
  float tot = 0.f;
#pragma unroll
  for (int i = 0; i < 8; ++i) tot += red[i];
  const float rs = rsqrtf(tot / (float)D + 1e-5f);
  for (int i = tid; i < D; i += 256)
    Y[(size_t)m * D + i] = f2bf(x[i] * rs * W[i]);
}

// ---------------- Tiled bf16 WMMA GEMM: C = alpha*A@B + beta*C ----------------
// A: [M,K] bf16 row-major, B: [K,N] bf16 row-major, C: [M,N] f32.
// K must be a multiple of 64 (true for all call sites).
#define TILE_M 128
#define TILE_N 128
#define TILE_K 64
#define LDSK   72   // padded K stride (bf16 elems): 144B rows, 16B-aligned chunks

__global__ __launch_bounds__(256) void gemm_bf16(
    const u16* __restrict__ A, const u16* __restrict__ B, float* __restrict__ C,
    int M, int N, int K, float alpha, float beta) {
  __shared__ u16 As[2][TILE_M * LDSK];
  __shared__ u16 Bs[2][TILE_N * LDSK];   // B transposed: Bs[n][k]
  const int tid  = threadIdx.x;
  const int lane = tid & 31;
  const int wave = tid >> 5;
  const int wm = wave >> 2, wn = wave & 3;        // 2 x 4 wave grid
  const int bm = blockIdx.y * TILE_M, bn = blockIdx.x * TILE_N;
  const int lm = lane & 15, lh = lane >> 4;
  const int nk = K / TILE_K;

  v8f acc[4][2] = {};

  // ---- prologue: stage tile 0 into buffer 0 ----
#if HAVE_TDM
  if (wave == 0)
    tdm_load_tile(A + (size_t)bm * K, lds_offset(&As[0][0]), K, M - bm, K);
#else
#pragma unroll
  for (int i = 0; i < 4; ++i) {
    int c = tid + 256 * i;
    int row = c >> 3, kc = (c & 7) * 8;
    *(uint4*)(&As[0][row * LDSK + kc]) =
        *(const uint4*)(A + (size_t)(bm + row) * K + kc);
  }
#endif
#pragma unroll
  for (int i = 0; i < 32; ++i) {
    int idx = i * 256 + tid;
    int kk = idx >> 7, nn = idx & 127;
    u16 d = 0;
    if (bn + nn < N) d = B[(size_t)kk * N + bn + nn];
    Bs[0][nn * LDSK + kk] = d;
  }
#if HAVE_TDM
  __builtin_amdgcn_s_wait_tensorcnt(0);
#endif
  __syncthreads();

  int buf = 0;
  for (int kt = 0; kt < nk; ++kt) {
    const int k0n = (kt + 1) * TILE_K;
    const bool pre = (kt + 1 < nk);
    u16 breg[32];
#if !HAVE_TDM
    uint4 areg[4];
#endif
    if (pre) {
      // async/prefetch next tile while this one computes
#if HAVE_TDM
      if (wave == 0)
        tdm_load_tile(A + (size_t)bm * K + k0n, lds_offset(&As[buf ^ 1][0]),
                      K - k0n, M - bm, K);
#else
#pragma unroll
      for (int i = 0; i < 4; ++i) {
        int c = tid + 256 * i;
        int row = c >> 3, kc = (c & 7) * 8;
        areg[i] = *(const uint4*)(A + (size_t)(bm + row) * K + k0n + kc);
      }
#endif
#pragma unroll
      for (int i = 0; i < 32; ++i) {
        int idx = i * 256 + tid;
        int kk = idx >> 7, nn = idx & 127;
        u16 d = 0;
        if (bn + nn < N) d = B[(size_t)(k0n + kk) * N + bn + nn];
        breg[i] = d;
      }
      if (kt + 2 < nk) {  // L2 warm-up for tile k+2
        __builtin_prefetch(A + (size_t)(bm + (tid >> 1)) * K +
                               (size_t)(kt + 2) * TILE_K + (tid & 1) * 32, 0, 1);
        __builtin_prefetch(B + (size_t)((kt + 2) * TILE_K + (tid >> 2)) * N +
                               bn + (tid & 3) * 32, 0, 1);
      }
    }
    // ---- compute from current buffer ----
    const u16* Ab = As[buf];
    const u16* Bb = Bs[buf];
#pragma unroll
    for (int kk2 = 0; kk2 < 2; ++kk2) {
      const int kb = kk2 * 32;
      Frag a[4], b[2];
#pragma unroll
      for (int i = 0; i < 4; ++i) {
        const u16* p = &Ab[(wm * 64 + i * 16 + lm) * LDSK + kb + lh * 8];
        a[i].q[0] = *(const uint4*)p;
        a[i].q[1] = *(const uint4*)(p + 16);
      }
#pragma unroll
      for (int j = 0; j < 2; ++j) {
        const u16* p = &Bb[(wn * 32 + j * 16 + lm) * LDSK + kb + lh * 8];
        b[j].q[0] = *(const uint4*)p;
        b[j].q[1] = *(const uint4*)(p + 16);
      }
#pragma unroll
      for (int i = 0; i < 4; ++i)
#pragma unroll
        for (int j = 0; j < 2; ++j)
          acc[i][j] = wmma_bf16(a[i], b[j], acc[i][j]);
    }
    // ---- publish next buffer ----
    if (pre) {
#if !HAVE_TDM
#pragma unroll
      for (int i = 0; i < 4; ++i) {
        int c = tid + 256 * i;
        int row = c >> 3, kc = (c & 7) * 8;
        *(uint4*)(&As[buf ^ 1][row * LDSK + kc]) = areg[i];
      }
#endif
#pragma unroll
      for (int i = 0; i < 32; ++i) {
        int idx = i * 256 + tid;
        int kk = idx >> 7, nn = idx & 127;
        Bs[buf ^ 1][nn * LDSK + kk] = breg[i];
      }
#if HAVE_TDM
      __builtin_amdgcn_s_wait_tensorcnt(0);
#endif
    }
    __syncthreads();
    buf ^= 1;
  }

  // ---- epilogue: C layout — lane holds col=lm, rows (lh*8 + v) ----
#pragma unroll
  for (int i = 0; i < 4; ++i)
#pragma unroll
    for (int j = 0; j < 2; ++j) {
      const int col = bn + wn * 32 + j * 16 + lm;
      if (col < N) {
#pragma unroll
        for (int v = 0; v < 8; ++v) {
          const int row = bm + wm * 64 + i * 16 + lh * 8 + v;
          if (row < M) {
            size_t off = (size_t)row * N + col;
            float r = alpha * acc[i][j][v];
            if (beta != 0.f) r += beta * C[off];
            C[off] = r;
          }
        }
      }
    }
}

// ---------------- RoPE + rank-16 LoRA fused, pack bf16 [B,nh,S,HD] ----------------
__global__ __launch_bounds__(128) void rope_pack(
    const float* __restrict__ X,   // [M, nh*HD] f32 base projection
    const float* __restrict__ T,   // [M, R]   f32  (x @ A)
    const float* __restrict__ Bm,  // [R, nh*HD] f32 LoRA B
    const float* __restrict__ Cos, const float* __restrict__ Sin,  // [S, HD]
    u16* __restrict__ Dst, int nh, int doRope) {
  __shared__ float sh[HDn];
  const int m = blockIdx.x;         // token
  const int h = blockIdx.y;
  const int d = threadIdx.x;
  const int b = m >> 10, s = m & (Sq - 1);
  const int NW = nh * HDn;
  const int col = h * HDn + d;
  float lo = 0.f;
#pragma unroll
  for (int r = 0; r < Rr; ++r) lo += T[(size_t)m * Rr + r] * Bm[(size_t)r * NW + col];
  float x = X[(size_t)m * NW + col] + SCALING * lo;
  float y = x;
  if (doRope) {
    sh[d] = x;
    __syncthreads();
    const float rot = (d < 64) ? -sh[d + 64] : sh[d - 64];
    y = x * Cos[s * HDn + d] + rot * Sin[s * HDn + d];
  }
  Dst[(((size_t)b * nh + h) * Sq + s) * HDn + d] = f2bf(y);
}

// ---------------- Flash attention (causal, GQA 2:1), WMMA ----------------
// grid: (B*H, S/64), block 128 (4 waves, each owns 16 query rows)
__global__ __launch_bounds__(128) void attn_kernel(
    const u16* __restrict__ Q,   // [B,H,S,HD] bf16
    const u16* __restrict__ Kc,  // [B,KVH,S,HD] bf16
    const u16* __restrict__ Vc,  // [B,KVH,S,HD] bf16
    float* __restrict__ O) {     // [M, H*HD] f32
  __shared__ u16 kls[32 * 136];        // K tile row-major, padded
  __shared__ u16 vls[HDn * 40];        // V tile transposed: [d][n], padded
  __shared__ u16 pls[4][16 * 40];      // per-wave P tile [m][n], padded
  const int bh = blockIdx.x;
  const int b = bh / Hn, h = bh % Hn, kvh = h >> 1;
  const int qt = blockIdx.y;
  const int qbase = qt * 64;
  const int tid = threadIdx.x, lane = tid & 31, wave = tid >> 5;
  const int lm = lane & 15, lh = lane >> 4;

  const u16* qp = Q  + ((size_t)bh * Sq + qbase + wave * 16) * HDn;
  const u16* kp = Kc + ((size_t)(b * KVHn + kvh) * Sq) * HDn;
  const u16* vp = Vc + ((size_t)(b * KVHn + kvh) * Sq) * HDn;

  // persistent Q fragments (A-layout), 4 k-steps of 32
  Frag qf[4];
#pragma unroll
  for (int ks = 0; ks < 4; ++ks) {
    const u16* p = qp + (size_t)lm * HDn + ks * 32 + lh * 8;
    qf[ks].q[0] = *(const uint4*)p;
    qf[ks].q[1] = *(const uint4*)(p + 16);
  }

  v8f oacc[8] = {};
  float mrow[8], lrow[8];
#pragma unroll
  for (int v = 0; v < 8; ++v) { mrow[v] = -1e30f; lrow[v] = 0.f; }

  const float rsc = 0.08838834764831845f;  // 1/sqrt(128)
  const int njt = qt * 2 + 2;              // 32-key tiles needed (causal)
  for (int jt = 0; jt < njt; ++jt) {
    const int jbase = jt * 32;
    __syncthreads();
    // stage K tile (32 x 128 bf16) row-major
#pragma unroll
    for (int i = 0; i < 4; ++i) {
      int c = tid + 128 * i;
      int row = c >> 4, dc = (c & 15) * 8;
      uint4 d = *(const uint4*)(kp + (size_t)(jbase + row) * HDn + dc);
      *(uint4*)(&kls[row * 136 + dc]) = d;
    }
    // stage V tile transposed: vls[d][n]
#pragma unroll
    for (int i = 0; i < 32; ++i) {
      int idx = i * 128 + tid;
      int row = idx >> 7, d = idx & 127;
      vls[d * 40 + row] = vp[(size_t)(jbase + row) * HDn + d];
    }
    __syncthreads();

    // S = Q K^T for 2 n-tiles of 16 keys
    v8f s0 = {}, s1 = {};
#pragma unroll
    for (int ks = 0; ks < 4; ++ks) {
      Frag b0, b1;
      const u16* p0 = &kls[lm * 136 + ks * 32 + lh * 8];
      b0.q[0] = *(const uint4*)p0; b0.q[1] = *(const uint4*)(p0 + 16);
      const u16* p1 = &kls[(16 + lm) * 136 + ks * 32 + lh * 8];
      b1.q[0] = *(const uint4*)p1; b1.q[1] = *(const uint4*)(p1 + 16);
      s0 = wmma_bf16(qf[ks], b0, s0);
      s1 = wmma_bf16(qf[ks], b1, s1);
    }

    const int qrow0 = qbase + wave * 16 + lh * 8;
#pragma unroll
    for (int v = 0; v < 8; ++v) {
      const int rq = qrow0 + v;
      float a0 = s0[v] * rsc + ((jbase + lm)      <= rq ? 0.f : -1e9f);
      float a1 = s1[v] * rsc + ((jbase + 16 + lm) <= rq ? 0.f : -1e9f);
      float mx = fmaxf(a0, a1);
#pragma unroll
      for (int off = 8; off >= 1; off >>= 1) mx = fmaxf(mx, __shfl_xor(mx, off, 32));
      const float mnew = fmaxf(mrow[v], mx);
      const float fsc  = __expf(mrow[v] - mnew);
      const float p0 = __expf(a0 - mnew);
      const float p1 = __expf(a1 - mnew);
      float sum = p0 + p1;
#pragma unroll
      for (int off = 8; off >= 1; off >>= 1) sum += __shfl_xor(sum, off, 32);
      lrow[v] = lrow[v] * fsc + sum;
      mrow[v] = mnew;
#pragma unroll
      for (int dt = 0; dt < 8; ++dt) oacc[dt][v] *= fsc;
      // C-layout -> LDS (P tile [m][n] bf16)
      pls[wave][(v + 8 * lh) * 40 + lm]      = f2bf(p0);
      pls[wave][(v + 8 * lh) * 40 + 16 + lm] = f2bf(p1);
    }
    asm volatile("s_wait_dscnt 0x0" ::: "memory");  // wave-local LDS RAW fence

    // P (A-layout) fragment
    Frag pf;
    const u16* pp = &pls[wave][lm * 40 + lh * 8];
    pf.q[0] = *(const uint4*)pp;
    pf.q[1] = *(const uint4*)(pp + 16);
    // O += P @ V  (8 d-tiles of 16)
#pragma unroll
    for (int dt = 0; dt < 8; ++dt) {
      Frag bv;
      const u16* vq = &vls[(dt * 16 + lm) * 40 + lh * 8];
      bv.q[0] = *(const uint4*)vq; bv.q[1] = *(const uint4*)(vq + 16);
      oacc[dt] = wmma_bf16(pf, bv, oacc[dt]);
    }
  }

  // epilogue: O[m][h*HD + d] = oacc / l
#pragma unroll
  for (int dt = 0; dt < 8; ++dt)
#pragma unroll
    for (int v = 0; v < 8; ++v) {
      const int rq = qbase + wave * 16 + lh * 8 + v;
      const size_t m = (size_t)b * Sq + rq;
      O[m * (Hn * HDn) + h * HDn + dt * 16 + lm] = oacc[dt][v] / lrow[v];
    }
}

// ---------------- residual + attn-out LoRA: data2 = data + proj + SC*(to@ob) ----
__global__ __launch_bounds__(256) void resid_lora(
    const float* __restrict__ data, const float* __restrict__ proj,
    const float* __restrict__ T, const float* __restrict__ Bm,
    float* __restrict__ out) {
  const size_t idx = (size_t)blockIdx.x * 256 + threadIdx.x;
  const int m = (int)(idx >> 11);        // D = 2048
  const int d = (int)(idx & (Dm - 1));
  float lo = 0.f;
#pragma unroll
  for (int r = 0; r < Rr; ++r) lo += T[(size_t)m * Rr + r] * Bm[(size_t)r * Dm + d];
  out[idx] = data[idx] + proj[idx] + SCALING * lo;
}

// ---------------- MoE gate: softmax + top-2 -> dense expert weights ----------------
__global__ __launch_bounds__(256) void gate_topk(
    const u16* __restrict__ sn, const float* __restrict__ gw,
    float* __restrict__ ew) {
  const int warp = threadIdx.x >> 5, lane = threadIdx.x & 31;
  const int m = blockIdx.x * 8 + warp;
  float acc[En] = {};
  const u16* x = sn + (size_t)m * Dm;
  for (int i = lane; i < Dm; i += 32) {
    const float v = bf2f(x[i]);
#pragma unroll
    for (int e = 0; e < En; ++e) acc[e] += v * gw[i * En + e];
  }
#pragma unroll
  for (int e = 0; e < En; ++e)
#pragma unroll
    for (int off = 16; off >= 1; off >>= 1) acc[e] += __shfl_xor(acc[e], off, 32);
  if (lane == 0) {
    float mx = acc[0];
#pragma unroll
    for (int e = 1; e < En; ++e) mx = fmaxf(mx, acc[e]);
    float p[En], sum = 0.f;
#pragma unroll
    for (int e = 0; e < En; ++e) { p[e] = __expf(acc[e] - mx); sum += p[e]; }
#pragma unroll
    for (int e = 0; e < En; ++e) p[e] /= sum;
    int i1 = 0; float v1 = p[0];
#pragma unroll
    for (int e = 1; e < En; ++e) if (p[e] > v1) { v1 = p[e]; i1 = e; }
    int i2 = -1; float v2 = -1.f;
#pragma unroll
    for (int e = 0; e < En; ++e) if (e != i1 && p[e] > v2) { v2 = p[e]; i2 = e; }
    const float tot = v1 + v2;
#pragma unroll
    for (int e = 0; e < En; ++e)
      ew[(size_t)m * En + e] = (e == i1) ? v1 / tot : ((e == i2) ? v2 / tot : 0.f);
  }
}

// ---------------- expert SiLU gate with fused rank-16 LoRA on w1/w3 ----------------
__global__ __launch_bounds__(256) void moe_silu(
    const float* __restrict__ cw1, const float* __restrict__ cw3,
    const float* __restrict__ t1, const float* __restrict__ t3,
    const float* __restrict__ b1, const float* __restrict__ b3,
    u16* __restrict__ sr) {
  const size_t idx = (size_t)blockIdx.x * 256 + threadIdx.x;
  const int m = (int)(idx >> 13);          // DFF = 8192
  const int f = (int)(idx & (DFFn - 1));
  float l1 = 0.f, l3 = 0.f;
#pragma unroll
  for (int r = 0; r < Rr; ++r) {
    l1 += t1[(size_t)m * Rr + r] * b1[(size_t)r * DFFn + f];
    l3 += t3[(size_t)m * Rr + r] * b3[(size_t)r * DFFn + f];
  }
  const float x1 = cw1[idx] + SCALING * l1;
  const float x3 = cw3[idx] + SCALING * l3;
  const float sv = x1 / (1.f + __expf(-x1));
  sr[idx] = f2bf(sv * x3);
}

// ---------------- expert accumulate: out += ew * (hs + SC*(t2@b2)) ----------------
__global__ __launch_bounds__(256) void moe_accum(
    float* __restrict__ out, const float* __restrict__ hs,
    const float* __restrict__ t2, const float* __restrict__ b2,
    const float* __restrict__ ew, int e) {
  const size_t idx = (size_t)blockIdx.x * 256 + threadIdx.x;
  const int m = (int)(idx >> 11);
  const int d = (int)(idx & (Dm - 1));
  float lo = 0.f;
#pragma unroll
  for (int r = 0; r < Rr; ++r) lo += t2[(size_t)m * Rr + r] * b2[(size_t)r * Dm + d];
  out[idx] += ew[(size_t)m * En + e] * (hs[idx] + SCALING * lo);
}

// ==================== host launch ====================
extern "C" void kernel_launch(void* const* d_in, const int* in_sizes, int n_in,
                              void* d_out, int out_size, void* d_ws, size_t ws_size,
                              hipStream_t stream) {
  (void)in_sizes; (void)n_in; (void)out_size; (void)ws_size;
  const float* data = (const float*)d_in[0];
  const float* cosT = (const float*)d_in[2];
  const float* sinT = (const float*)d_in[3];
  const float* anw  = (const float*)d_in[4];
  const float* fnw  = (const float*)d_in[5];
  const float* wq   = (const float*)d_in[6];
  const float* wk   = (const float*)d_in[7];
  const float* wv   = (const float*)d_in[8];
  const float* wo   = (const float*)d_in[9];
  const float* qa   = (const float*)d_in[10];
  const float* qb   = (const float*)d_in[11];
  const float* ka   = (const float*)d_in[12];
  const float* kb   = (const float*)d_in[13];
  const float* va   = (const float*)d_in[14];
  const float* vb   = (const float*)d_in[15];
  const float* oa   = (const float*)d_in[16];
  const float* ob   = (const float*)d_in[17];
  const float* gw   = (const float*)d_in[18];
  const float* w1   = (const float*)d_in[19];
  const float* w2   = (const float*)d_in[20];
  const float* w3   = (const float*)d_in[21];
  const float* a1   = (const float*)d_in[22];
  const float* b1   = (const float*)d_in[23];
  const float* a3   = (const float*)d_in[24];
  const float* b3   = (const float*)d_in[25];
  const float* a2   = (const float*)d_in[26];
  const float* b2   = (const float*)d_in[27];
  float* out = (float*)d_out;

  const int M = Mtok;
  char* wp = (char*)d_ws;
  auto alloc = [&](size_t bytes) -> void* {
    void* p = (void*)wp; wp += (bytes + 255) & ~(size_t)255; return p;
  };

  u16* h_bf   = (u16*)alloc((size_t)M * Dm * 2);
  u16* wq_bf  = (u16*)alloc((size_t)Dm * (Hn * HDn) * 2);
  u16* wk_bf  = (u16*)alloc((size_t)Dm * (KVHn * HDn) * 2);
  u16* wv_bf  = (u16*)alloc((size_t)Dm * (KVHn * HDn) * 2);
  u16* wo_bf  = (u16*)alloc((size_t)(Hn * HDn) * Dm * 2);
  u16* qa_bf  = (u16*)alloc((size_t)Dm * Rr * 2);
  u16* ka_bf  = (u16*)alloc((size_t)Dm * Rr * 2);
  u16* va_bf  = (u16*)alloc((size_t)Dm * Rr * 2);
  u16* oa_bf  = (u16*)alloc((size_t)(Hn * HDn) * Rr * 2);
  u16* a1_bf  = (u16*)alloc((size_t)En * Dm * Rr * 2);
  u16* a3_bf  = (u16*)alloc((size_t)En * Dm * Rr * 2);
  u16* a2_bf  = (u16*)alloc((size_t)En * DFFn * Rr * 2);
  u16* w1_bf  = (u16*)alloc((size_t)Dm * DFFn * 2);
  u16* w3_bf  = (u16*)alloc((size_t)Dm * DFFn * 2);
  u16* w2_bf  = (u16*)alloc((size_t)DFFn * Dm * 2);
  float* xq   = (float*)alloc((size_t)M * (Hn * HDn) * 4);
  float* xk   = (float*)alloc((size_t)M * (KVHn * HDn) * 4);
  float* xv   = (float*)alloc((size_t)M * (KVHn * HDn) * 4);
  float* tq   = (float*)alloc((size_t)M * Rr * 4);
  float* tk   = (float*)alloc((size_t)M * Rr * 4);
  float* tv   = (float*)alloc((size_t)M * Rr * 4);
  u16* qpack  = (u16*)alloc((size_t)Bsz * Hn * Sq * HDn * 2);
  u16* kpack  = (u16*)alloc((size_t)Bsz * KVHn * Sq * HDn * 2);
  u16* vpack  = (u16*)alloc((size_t)Bsz * KVHn * Sq * HDn * 2);
  float* of32 = (float*)alloc((size_t)M * (Hn * HDn) * 4);
  u16* o_bf   = (u16*)alloc((size_t)M * (Hn * HDn) * 2);
  float* to   = (float*)alloc((size_t)M * Rr * 4);
  float* obuf = (float*)alloc((size_t)M * Dm * 4);
  float* data2= (float*)alloc((size_t)M * Dm * 4);
  u16* sn_bf  = (u16*)alloc((size_t)M * Dm * 2);
  float* cw1  = (float*)alloc((size_t)M * DFFn * 4);
  float* cw3  = (float*)alloc((size_t)M * DFFn * 4);
  float* ewts = (float*)alloc((size_t)M * En * 4);
  float* t1   = (float*)alloc((size_t)M * Rr * 4);
  float* t3   = (float*)alloc((size_t)M * Rr * 4);
  float* t2   = (float*)alloc((size_t)M * Rr * 4);
  u16* sr_bf  = (u16*)alloc((size_t)M * DFFn * 2);
  float* hsf  = (float*)alloc((size_t)M * Dm * 4);

  auto cvt = [&](const float* src, u16* dst, size_t n) {
    cvt_bf16<<<dim3(2048), dim3(256), 0, stream>>>(src, dst, n);
  };
  auto gemm = [&](const u16* A, const u16* Bm_, float* C, int Md, int Nd, int Kd,
                  float alpha, float beta) {
    dim3 g((unsigned)((Nd + TILE_N - 1) / TILE_N), (unsigned)((Md + TILE_M - 1) / TILE_M));
    gemm_bf16<<<g, dim3(256), 0, stream>>>(A, Bm_, C, Md, Nd, Kd, alpha, beta);
  };

  // ---- weight conversions (once per launch; ~130MB bf16, L2-resident) ----
  cvt(wq, wq_bf, (size_t)Dm * Hn * HDn);
  cvt(wk, wk_bf, (size_t)Dm * KVHn * HDn);
  cvt(wv, wv_bf, (size_t)Dm * KVHn * HDn);
  cvt(wo, wo_bf, (size_t)Hn * HDn * Dm);
  cvt(qa, qa_bf, (size_t)Dm * Rr);
  cvt(ka, ka_bf, (size_t)Dm * Rr);
  cvt(va, va_bf, (size_t)Dm * Rr);
  cvt(oa, oa_bf, (size_t)Hn * HDn * Rr);
  cvt(a1, a1_bf, (size_t)En * Dm * Rr);
  cvt(a3, a3_bf, (size_t)En * Dm * Rr);
  cvt(a2, a2_bf, (size_t)En * DFFn * Rr);
  cvt(w1, w1_bf, (size_t)Dm * DFFn);
  cvt(w3, w3_bf, (size_t)Dm * DFFn);
  cvt(w2, w2_bf, (size_t)DFFn * Dm);

  // ---- attention block ----
  rmsnorm_bf16<<<dim3(M), dim3(256), 0, stream>>>(data, anw, h_bf, Dm);

  gemm(h_bf, wq_bf, xq, M, Hn * HDn,  Dm, 1.f, 0.f);
  gemm(h_bf, wk_bf, xk, M, KVHn * HDn, Dm, 1.f, 0.f);
  gemm(h_bf, wv_bf, xv, M, KVHn * HDn, Dm, 1.f, 0.f);
  gemm(h_bf, qa_bf, tq, M, Rr, Dm, 1.f, 0.f);
  gemm(h_bf, ka_bf, tk, M, Rr, Dm, 1.f, 0.f);
  gemm(h_bf, va_bf, tv, M, Rr, Dm, 1.f, 0.f);

  rope_pack<<<dim3(M, Hn),   dim3(HDn), 0, stream>>>(xq, tq, qb, cosT, sinT, qpack, Hn, 1);
  rope_pack<<<dim3(M, KVHn), dim3(HDn), 0, stream>>>(xk, tk, kb, cosT, sinT, kpack, KVHn, 1);
  rope_pack<<<dim3(M, KVHn), dim3(HDn), 0, stream>>>(xv, tv, vb, cosT, sinT, vpack, KVHn, 0);

  attn_kernel<<<dim3(Bsz * Hn, Sq / 64), dim3(128), 0, stream>>>(qpack, kpack, vpack, of32);
  cvt(of32, o_bf, (size_t)M * Hn * HDn);

  gemm(o_bf, wo_bf, obuf, M, Dm, Hn * HDn, 1.f, 0.f);
  gemm(o_bf, oa_bf, to,   M, Rr, Hn * HDn, 1.f, 0.f);
  resid_lora<<<dim3((M * Dm) / 256), dim3(256), 0, stream>>>(data, obuf, to, ob, data2);

  // ---- MoE block ----
  rmsnorm_bf16<<<dim3(M), dim3(256), 0, stream>>>(data2, fnw, sn_bf, Dm);
  gemm(sn_bf, w1_bf, cw1, M, DFFn, Dm, 1.f, 0.f);
  gemm(sn_bf, w3_bf, cw3, M, DFFn, Dm, 1.f, 0.f);
  gate_topk<<<dim3(M / 8), dim3(256), 0, stream>>>(sn_bf, gw, ewts);

  copy_f32<<<dim3(2048), dim3(256), 0, stream>>>(data2, out, (size_t)M * Dm);

  for (int e = 0; e < En; ++e) {
    const u16* a1e = a1_bf + (size_t)e * Dm * Rr;
    const u16* a3e = a3_bf + (size_t)e * Dm * Rr;
    const u16* a2e = a2_bf + (size_t)e * DFFn * Rr;
    const float* b1e = b1 + (size_t)e * Rr * DFFn;
    const float* b3e = b3 + (size_t)e * Rr * DFFn;
    const float* b2e = b2 + (size_t)e * Rr * Dm;

    gemm(sn_bf, a1e, t1, M, Rr, Dm, 1.f, 0.f);
    gemm(sn_bf, a3e, t3, M, Rr, Dm, 1.f, 0.f);
    moe_silu<<<dim3((M * DFFn) / 256), dim3(256), 0, stream>>>(cw1, cw3, t1, t3, b1e, b3e, sr_bf);
    gemm(sr_bf, w2_bf, hsf, M, Dm, DFFn, 1.f, 0.f);
    gemm(sr_bf, a2e,  t2,  M, Rr, DFFn, 1.f, 0.f);
    moe_accum<<<dim3((M * Dm) / 256), dim3(256), 0, stream>>>(out, hsf, t2, b2e, ewts, e);
  }
}
